// Wordwise_79010218377317
// MI455X (gfx1250) — compile-verified, matrix operand order checked by reading, not run on hardware
//
#include <hip/hip_runtime.h>

// ---------------------------------------------------------------------------
// Wordwise pipeline for MI455X (gfx1250), round 2.
// Compute-bound (~390 GFLOP vs ~300MB traffic) -> all convs are implicit GEMM
// on v_wmma_f32_16x16x32_bf16. This round restructures data layouts so every
// WMMA fragment is fed by wide 16B loads:
//   - activations channel-last  -> B fragments = 2 x ds_load_b128
//   - weights pre-transposed    -> A fragments = 2 x global_load_b128
//   - results packed            -> 1 x global_store_b128 per lane
// ---------------------------------------------------------------------------

typedef __bf16 bf16;
typedef __attribute__((ext_vector_type(16))) __bf16 v16bf;
typedef __attribute__((ext_vector_type(8)))  __bf16 v8bf;
typedef __attribute__((ext_vector_type(8)))  float  v8f;

constexpr int B_    = 32;
constexpr int NMELS = 80;
constexpr int CIP1  = 96;   // NMELS padded to a multiple of 32
constexpr int T_    = 4096;
constexpr int HID   = 128;
constexpr int MW    = 64;   // MAX_WORDS
constexpr int MD    = 100;  // MAX_DUR

__device__ __forceinline__ v8bf zero8() {
    v8bf v;
#pragma unroll
    for (int i = 0; i < 8; ++i) v[i] = (bf16)0.0f;
    return v;
}

// 16-bit WMMA A/B fragment (ISA 05_wmma 7.12.2): lane holds K slots
// {off..off+7, off+16..off+23} with off = (lane&16)?8:0. With K contiguous
// in memory this is two 16-byte loads.
__device__ __forceinline__ v16bf ldfrag(const bf16* p) {
    v8bf lo = *(const v8bf*)p;        // slots 0..7
    v8bf hi = *(const v8bf*)(p + 16); // slots 8..15
    return __builtin_shufflevector(lo, hi, 0, 1, 2, 3, 4, 5, 6, 7,
                                   8, 9, 10, 11, 12, 13, 14, 15);
}

// --------------- features f32 [B,NMELS,T] -> bf16 [B,T,CIP1] ---------------
__global__ void cvt_features(const float* __restrict__ src, bf16* __restrict__ dst) {
    size_t i = (size_t)blockIdx.x * blockDim.x + threadIdx.x;
    if (i >= (size_t)B_ * T_ * CIP1) return;
    int ci = (int)(i % CIP1);
    size_t r = i / CIP1;
    int t = (int)(r % T_);
    int b = (int)(r / T_);
    dst[i] = (ci < NMELS) ? (bf16)src[((size_t)b * NMELS + ci) * T_ + t]
                          : (bf16)0.0f;
}

// --------------- weights f32 [HID,CIN,NTAP] -> bf16 [NTAP,HID,CIP] ---------
__global__ void xpose_w(const float* __restrict__ src, bf16* __restrict__ dst,
                        int CIN, int CIPad, int NTAP) {
    int i = blockIdx.x * blockDim.x + threadIdx.x;
    if (i >= NTAP * HID * CIPad) return;
    int ci = i % CIPad;
    int r  = i / CIPad;
    int co = r % HID;
    int tap = r / HID;
    dst[i] = (ci < CIN) ? (bf16)src[((size_t)co * CIN + ci) * NTAP + tap]
                        : (bf16)0.0f;
}

// ------------------------------ conv1d (SAME, k=5) -------------------------
// y[b,t,co] = relu?( bias[co] + sum_{ci,tap} wt[tap,co,ci] * x[b,t+tap-2,ci] )
// per block: one 16-wide t tile, 8 waves cover co 0..127.
template <int CIP>
__global__ void conv1d_wmma(const bf16* __restrict__ x,     // [B,T,CIP]
                            const bf16* __restrict__ wt,    // [5,HID,CIP]
                            const float* __restrict__ bias, // [HID]
                            bf16* __restrict__ y,           // [B,T,HID]
                            int relu) {
    __shared__ bf16 sx[20 * CIP];                // halo rows t0-2 .. t0+17
    constexpr int ROWV = CIP / 8;
    const int b = blockIdx.y, t0 = blockIdx.x * 16, tid = threadIdx.x;

    for (int e = tid; e < 20 * ROWV; e += 256) { // pure b128 copies
        int j = e / ROWV, cv = e - j * ROWV;
        int t = t0 - 2 + j;
        v8bf v = zero8();
        if (t >= 0 && t < T_)
            v = *(const v8bf*)&x[((size_t)b * T_ + t) * CIP + cv * 8];
        *(v8bf*)&sx[j * CIP + cv * 8] = v;
    }
    __syncthreads();

    const int wave = tid >> 5, lane = tid & 31;
    const int co_base = wave * 16;
    const int row = lane & 15, col = lane & 15;
    const int off = (lane & 16) ? 8 : 0;

    v8f acc = {};
    for (int tap = 0; tap < 5; ++tap) {
        const bf16* wrow = wt + ((size_t)tap * HID + co_base + row) * CIP;
        const bf16* brow = &sx[(col + tap) * CIP];
#pragma unroll
        for (int c = 0; c < CIP / 32; ++c) {
            int k = c * 32 + off;
            v16bf a  = ldfrag(wrow + k);   // 2 x global_load_b128
            v16bf bm = ldfrag(brow + k);   // 2 x ds_load_b128
            acc = __builtin_amdgcn_wmma_f32_16x16x32_bf16(
                false, a, false, bm, (short)0, acc, false, false);
        }
    }

    const int co0 = co_base + ((lane & 16) >> 1);   // lanes>=16 -> rows M+8
    v8bf o;
#pragma unroll
    for (int i = 0; i < 8; ++i) {
        float v = acc[i] + bias[co0 + i];
        if (relu) v = fmaxf(v, 0.0f);
        o[i] = (bf16)v;
    }
    *(v8bf*)&y[((size_t)b * T_ + t0 + col) * HID + co0] = o; // b128 store
}

// ------------------------------ gather_words -------------------------------
// g[b,w,d,c] = mask ? emb[b,start+d,c] : 0      (vectorized over c by 8)
__global__ void gather_kernel(const bf16* __restrict__ emb, // [B,T,HID]
                              const int* __restrict__ wb,   // [B,MW,2]
                              bf16* __restrict__ g) {       // [B,MW,MD,HID]
    size_t i = (size_t)blockIdx.x * blockDim.x + threadIdx.x;
    if (i >= (size_t)B_ * MW * MD * (HID / 8)) return;
    int cv = (int)(i % (HID / 8));
    size_t r = i / (HID / 8);
    int d = (int)(r % MD); r /= MD;
    int w = (int)(r % MW);
    int b = (int)(r / MW);
    int start = wb[(b * MW + w) * 2 + 0];
    int end   = wb[(b * MW + w) * 2 + 1];
    int idx = start + d;
    v8bf v = zero8();
    if (d < (end - start) && idx < T_)
        v = *(const v8bf*)&emb[((size_t)b * T_ + idx) * HID + cv * 8];
    *(v8bf*)&g[(((size_t)b * MW + w) * MD + d) * HID + cv * 8] = v;
}

// ------------------------------ conv2d 5x5 (SAME) --------------------------
// y[b,w,d,co] = relu( bias[co] + sum_{ci,kh,kw} wt[kh*5+kw,co,ci] *
//                     x[b,w+kh-2,d+kw-2,ci] )   on the 64x100 grid.
// Per block: one w row, 32-wide d tile; each wave: 16 co x (2 x 16 d) tiles,
// A fragment shared between the two d tiles.
__global__ void conv2d_wmma(const bf16* __restrict__ x,     // [B,MW,MD,HID]
                            const bf16* __restrict__ wt,    // [25,HID,HID]
                            const float* __restrict__ bias, // [HID]
                            bf16* __restrict__ y) {         // [B,MW,MD,HID]
    __shared__ bf16 sx[5 * 36 * HID];            // [kh][d halo][ci]: 46 KB
    const int b = blockIdx.z, wpos = blockIdx.y, d0 = blockIdx.x * 32;
    const int tid = threadIdx.x;

    for (int e = tid; e < 5 * 36 * (HID / 8); e += 256) { // b128 copies
        int cv = e & 15;
        int r  = e >> 4;
        int j = r % 36, wi5 = r / 36;
        int wi = wpos + wi5 - 2, d = d0 - 2 + j;
        v8bf v = zero8();
        if (wi >= 0 && wi < MW && (unsigned)d < (unsigned)MD)
            v = *(const v8bf*)&x[(((size_t)b * MW + wi) * MD + d) * HID + cv * 8];
        *(v8bf*)&sx[((size_t)wi5 * 36 + j) * HID + cv * 8] = v;
    }
    __syncthreads();

    const int wave = tid >> 5, lane = tid & 31;
    const int co_base = wave * 16;
    const int row = lane & 15, col = lane & 15;
    const int off = (lane & 16) ? 8 : 0;

    v8f acc0 = {}, acc1 = {};
    for (int kh = 0; kh < 5; ++kh) {
        for (int kw = 0; kw < 5; ++kw) {
            const int tap = kh * 5 + kw;
            const bf16* wrow = wt + ((size_t)tap * HID + co_base + row) * HID;
            if (tap < 24)                          // warm next tap's row (L2)
                __builtin_prefetch(wrow + HID * HID, 0, 1);
            const bf16* br0 = &sx[((size_t)kh * 36 + col + kw) * HID];
            const bf16* br1 = br0 + 16 * HID;
#pragma unroll
            for (int c = 0; c < 4; ++c) {
                int k = c * 32 + off;
                v16bf a  = ldfrag(wrow + k);       // shared by both d tiles
                v16bf b0 = ldfrag(br0 + k);
                v16bf b1 = ldfrag(br1 + k);
                acc0 = __builtin_amdgcn_wmma_f32_16x16x32_bf16(
                    false, a, false, b0, (short)0, acc0, false, false);
                acc1 = __builtin_amdgcn_wmma_f32_16x16x32_bf16(
                    false, a, false, b1, (short)0, acc1, false, false);
            }
        }
    }

    const int co0 = co_base + ((lane & 16) >> 1);
    const int d1 = d0 + col, d2 = d0 + 16 + col;
    if (d1 < MD) {
        v8bf o;
#pragma unroll
        for (int i = 0; i < 8; ++i)
            o[i] = (bf16)fmaxf(acc0[i] + bias[co0 + i], 0.0f);
        *(v8bf*)&y[(((size_t)b * MW + wpos) * MD + d1) * HID + co0] = o;
    }
    if (d2 < MD) {
        v8bf o;
#pragma unroll
        for (int i = 0; i < 8; ++i)
            o[i] = (bf16)fmaxf(acc1[i] + bias[co0 + i], 0.0f);
        *(v8bf*)&y[(((size_t)b * MW + wpos) * MD + d2) * HID + co0] = o;
    }
}

// ------------------------------ einsum + relu ------------------------------
// e[b,w,c] = relu( lb + sum_d x[b,w,d,c] * lw[d] )   (mem-bound, 52 MFLOP)
__global__ void einsum_kernel(const bf16* __restrict__ x,   // [B,MW,MD,HID]
                              const float* __restrict__ lw, // [MD]
                              const float* __restrict__ lb,
                              float* __restrict__ e) {      // [B,MW,HID]
    int i = blockIdx.x * blockDim.x + threadIdx.x;
    if (i >= B_ * MW * HID) return;
    int c = i % HID;
    int r = i / HID;
    int w = r % MW, b = r / MW;
    const bf16* p = x + (((size_t)b * MW + w) * MD) * HID + c;
    float s = lb[0];
#pragma unroll 4
    for (int d = 0; d < MD; ++d) s += (float)p[(size_t)d * HID] * lw[d];
    e[i] = fmaxf(s, 0.0f);
}

// ------------------------------ final conv (degenerate W=1) ----------------
// out[b,w] = b6 + sum_{kh,c} w6[0,c,kh,2] * e[b,w+kh-2,c]
__global__ void conv6_kernel(const float* __restrict__ e,   // [B,MW,HID]
                             const float* __restrict__ w6,  // [1,HID,5,5] f32
                             const float* __restrict__ b6,
                             float* __restrict__ out) {     // [B,1,MW,1]
    int i = blockIdx.x * blockDim.x + threadIdx.x;
    if (i >= B_ * MW) return;
    int w = i % MW, b = i / MW;
    float s = b6[0];
#pragma unroll
    for (int kh = 0; kh < 5; ++kh) {
        int wi = w + kh - 2;
        if (wi < 0 || wi >= MW) continue;
        const float* ep = e + ((size_t)b * MW + wi) * HID;
        for (int c = 0; c < HID; ++c)
            s += w6[c * 25 + kh * 5 + 2] * ep[c];
    }
    out[i] = s;
}

// ---------------------------------------------------------------------------
extern "C" void kernel_launch(void* const* d_in, const int* in_sizes, int n_in,
                              void* d_out, int out_size, void* d_ws, size_t ws_size,
                              hipStream_t stream) {
    const float* features = (const float*)d_in[0];
    const int*   wb       = (const int*)d_in[1];
    const float* w1 = (const float*)d_in[2];
    const float* b1 = (const float*)d_in[3];
    const float* w2 = (const float*)d_in[4];
    const float* b2 = (const float*)d_in[5];
    const float* w3 = (const float*)d_in[6];
    const float* b3 = (const float*)d_in[7];
    const float* w4 = (const float*)d_in[8];
    const float* b4 = (const float*)d_in[9];
    const float* w5 = (const float*)d_in[10];
    const float* b5 = (const float*)d_in[11];
    const float* lw = (const float*)d_in[12];
    const float* lb = (const float*)d_in[13];
    const float* w6 = (const float*)d_in[14];
    const float* b6 = (const float*)d_in[15];
    float* out = (float*)d_out;

    // --- carve scratch (all activations channel-last bf16) ---
    char* ws = (char*)d_ws;
    size_t o = 0;
    auto carve = [&](size_t bytes) {
        char* p = ws + o; o += (bytes + 255) & ~(size_t)255; return p;
    };
    bf16* xbf  = (bf16*)carve((size_t)B_ * T_ * CIP1 * 2);       // [B,T,96]
    bf16* h1   = (bf16*)carve((size_t)B_ * T_ * HID * 2);        // [B,T,128]
    bf16* h2   = (bf16*)carve((size_t)B_ * T_ * HID * 2);
    bf16* embb = (bf16*)carve((size_t)B_ * T_ * HID * 2);
    bf16* g    = (bf16*)carve((size_t)B_ * MW * MD * HID * 2);   // [B,MW,MD,128]
    bf16* y4   = (bf16*)carve((size_t)B_ * MW * MD * HID * 2);
    bf16* y5   = (bf16*)carve((size_t)B_ * MW * MD * HID * 2);
    float* ein = (float*)carve((size_t)B_ * MW * HID * 4);       // [B,MW,128]
    bf16* wt1  = (bf16*)carve((size_t)5 * HID * CIP1 * 2);       // [5,128,96]
    bf16* wt2  = (bf16*)carve((size_t)5 * HID * HID * 2);
    bf16* wt3  = (bf16*)carve((size_t)5 * HID * HID * 2);
    bf16* wt4  = (bf16*)carve((size_t)25 * HID * HID * 2);       // [25,128,128]
    bf16* wt5  = (bf16*)carve((size_t)25 * HID * HID * 2);

    // one-time layout conversions (memory-bound, small vs main convs)
    {
        size_t n = (size_t)B_ * T_ * CIP1;
        cvt_features<<<(unsigned)((n + 255) / 256), 256, 0, stream>>>(features, xbf);
    }
    auto xp = [&](const float* s, bf16* d, int cin, int cip, int ntap) {
        int n = ntap * HID * cip;
        xpose_w<<<(n + 255) / 256, 256, 0, stream>>>(s, d, cin, cip, ntap);
    };
    xp(w1, wt1, NMELS, CIP1, 5);
    xp(w2, wt2, HID, HID, 5);
    xp(w3, wt3, HID, HID, 5);
    xp(w4, wt4, HID, HID, 25);
    xp(w5, wt5, HID, HID, 25);

    // conv1d stack (8 waves/block cover all 128 output channels)
    conv1d_wmma<CIP1><<<dim3(T_ / 16, B_), 256, 0, stream>>>(xbf, wt1, b1, h1, 1);
    conv1d_wmma<HID> <<<dim3(T_ / 16, B_), 256, 0, stream>>>(h1,  wt2, b2, h2, 1);
    conv1d_wmma<HID> <<<dim3(T_ / 16, B_), 256, 0, stream>>>(h2,  wt3, b3, embb, 0);

    // word gather -> padded (B,MW,MD,HID), 16B per thread
    {
        size_t n = (size_t)B_ * MW * MD * (HID / 8);
        gather_kernel<<<(unsigned)((n + 255) / 256), 256, 0, stream>>>(embb, wb, g);
    }

    // 2D conv stack (dominant cost: 2 x ~168 GFLOP)
    conv2d_wmma<<<dim3((MD + 31) / 32, MW, B_), 256, 0, stream>>>(g,  wt4, b4, y4);
    conv2d_wmma<<<dim3((MD + 31) / 32, MW, B_), 256, 0, stream>>>(y4, wt5, b5, y5);

    // duration projection + relu, then final degenerate conv
    einsum_kernel<<<(B_ * MW * HID) / 256, 256, 0, stream>>>(y5, lw, lb, ein);
    conv6_kernel<<<(B_ * MW + 255) / 256, 256, 0, stream>>>(ein, w6, b6, out);
}
